// DG_Block_66151086293217
// MI455X (gfx1250) — matrix operand order repaired and checked by reading, not compiled, not run on hardware
//
#include <hip/hip_runtime.h>
#include <cstdint>

typedef __attribute__((ext_vector_type(16))) __bf16          v16bf;
typedef __attribute__((ext_vector_type(8)))  float           v8f;
typedef __attribute__((ext_vector_type(16))) unsigned short  v16us;
typedef __attribute__((ext_vector_type(8)))  unsigned short  v8us;
typedef __attribute__((ext_vector_type(4)))  int             v4i;

#define BB   8
#define CC   128
#define NN   2000
#define NPAD 2048
#define KNN  9
#define PSTRIDE 144   // ushorts per staged column (128 data + 16 pad; 288B, 32B-aligned)

#if defined(__has_builtin)
#if __has_builtin(__builtin_amdgcn_global_load_async_to_lds_b128) && \
    __has_builtin(__builtin_amdgcn_s_wait_asynccnt)
#define HAVE_ASYNC 1
#endif
#endif
#ifndef HAVE_ASYNC
#define HAVE_ASYNC 0
#endif

#define GAS __attribute__((address_space(1)))
#define LAS __attribute__((address_space(3)))

__device__ __forceinline__ unsigned short f2bf(float f) {
  unsigned u = __builtin_bit_cast(unsigned, f);
  u += 0x7FFFu + ((u >> 16) & 1u);            // round-to-nearest-even
  return (unsigned short)(u >> 16);
}
__device__ __forceinline__ v16bf as_bf16(v16us v) { return __builtin_bit_cast(v16bf, v); }

// Stage one 64-col x 128-ch bf16 panel (contiguous 16KB in global) into LDS
// with a 16-ushort per-column pad. Async path: 4 x b128 per wave, ASYNCcnt +4.
__device__ __forceinline__ void stage_panel(const unsigned short* __restrict__ gsrc,
                                            unsigned short* __restrict__ lbuf, int tid) {
#pragma unroll
  for (int e = 0; e < 4; ++e) {
    int chunk = tid + e * 256;            // 1024 chunks of 8 ushorts (16B)
    int r = chunk >> 4, c = chunk & 15;   // 16 chunks per column-row
    const unsigned short* gp = gsrc + r * 128 + c * 8;
    unsigned short* lp = lbuf + r * PSTRIDE + c * 8;
#if HAVE_ASYNC
    __builtin_amdgcn_global_load_async_to_lds_b128(
        (GAS v4i*)(uintptr_t)gp,
        (LAS v4i*)(unsigned)(uintptr_t)lp,
        0, 0);
#else
    *(v8us*)lp = *(const v8us*)gp;
#endif
  }
}

// ---------------------------------------------------------------------------
// prep: transpose (B,C,N) f32 -> [a][b][n][c] bf16 (point-major), plus ||x||^2.
// Padded points get zero vectors and sq=+huge so pd = -inf (never top-k'd).
// ---------------------------------------------------------------------------
__global__ void prep_kernel(const float* __restrict__ x, const float* __restrict__ m,
                            unsigned short* __restrict__ xb, float* __restrict__ sq) {
  int pid = blockIdx.x * blockDim.x + threadIdx.x;   // 2*8*2048 total
  int a = pid / (BB * NPAD);
  int r = pid % (BB * NPAD);
  int b = r / NPAD, n = r % NPAD;
  const float* src = a ? m : x;
  unsigned short* dst = xb + ((size_t)(a * BB + b) * NPAD + n) * CC;
  if (n < NN) {
    float s = 0.f;
    for (int c = 0; c < CC; ++c) {
      float v = src[(b * CC + c) * NN + n];
      s += v * v;
      dst[c] = f2bf(v);
    }
    sq[(a * BB + b) * NPAD + n] = s;
  } else {
    for (int c = 0; c < CC; ++c) dst[c] = 0;
    sq[(a * BB + b) * NPAD + n] = 1e30f;
  }
}

// ---------------------------------------------------------------------------
// Pack conv1 weights (C, 2C, 1, 3) into per-lane A-fragment blobs:
// blob[((br*3+dk)*8+cib)*8+cotile][lane][j], j->K = 16*(lane/16) + j (contiguous; B panels use the same order).
// ---------------------------------------------------------------------------
__global__ void wprep1_kernel(const float* __restrict__ w0, const float* __restrict__ w1,
                              unsigned short* __restrict__ blob) {
  int flat = blockIdx.x * 256 + threadIdx.x;          // 2*3*8*8*512 = 196608
  int j = flat & 15;  int t = flat >> 4;
  int lane = t & 31;  t >>= 5;
  int cot = t & 7;    t >>= 3;
  int cib = t & 7;    t >>= 3;
  int dk = t % 3;     int br = t / 3;
  int kk = j + ((lane < 16) ? 0 : 16);
  int ci = cib * 32 + kk;
  int co = cot * 16 + (lane & 15);
  const float* w = br ? w1 : w0;
  blob[flat] = f2bf(w[(co * 2 * CC + ci) * 3 + dk]);
}

// conv2 weights (C, C, 1, 3); K index = dk*128 + ci, 12 chunks of 32.
__global__ void wprep2_kernel(const float* __restrict__ w0, const float* __restrict__ w1,
                              unsigned short* __restrict__ blob) {
  int flat = blockIdx.x * 256 + threadIdx.x;          // 2*12*8*512 = 98304
  int j = flat & 15;  int t = flat >> 4;
  int lane = t & 31;  t >>= 5;
  int cot = t & 7;    t >>= 3;
  int s = t % 12;     int br = t / 12;
  int kk = j + ((lane < 16) ? 0 : 16);
  int k = s * 32 + kk;
  int dk = k >> 7, ci = k & 127;
  int co = cot * 16 + (lane & 15);
  const float* w = br ? w1 : w0;
  blob[flat] = f2bf(w[(co * CC + ci) * 3 + dk]);
}

// ---------------------------------------------------------------------------
// Fused Gram + top-9: each block owns 128 rows of one (array, batch) Gram
// matrix. B panels (64 cols x 128 ch bf16) are double-buffered in LDS via
// async global->LDS copies (ASYNCcnt) so panel cc+1 streams in while WMMA
// consumes panel cc. The 2000x2000 distance matrix never leaves the WGP.
// ---------------------------------------------------------------------------
__global__ void __launch_bounds__(256) gram_topk_kernel(const unsigned short* __restrict__ xb,
                                                        const float* __restrict__ sq,
                                                        int* __restrict__ idxout) {
  __shared__ unsigned short panel[2][64 * PSTRIDE];   // 2 x 18KB
  __shared__ float pdT[128 * 65];
  __shared__ float sqrow[128];
  __shared__ float sqcol[64];
  int tid = threadIdx.x, wave = tid >> 5, lane = tid & 31;
  int bx = blockIdx.x;
  int rb = bx & 15, b = (bx >> 4) & 7, a = bx >> 7;
  size_t abase = (size_t)(a * BB + b) * NPAD;
  int row0 = rb * 128;
  if (tid < 128) sqrow[tid] = sq[abase + row0 + tid];
  float tv[KNN]; int ti[KNN];
  for (int k = 0; k < KNN; ++k) { tv[k] = -3.0e38f; ti[k] = 0; }
  int myrow = row0 + wave * 16 + (lane & 15);
  const unsigned short* arow = xb + (abase + myrow) * CC;
  const unsigned short* gpanel = xb + abase * CC;
  int o0 = (lane < 16) ? 0 : 16;     // A: K 0..15 vs 16..31 contiguous
  int bo = (lane < 16) ? 0 : 16;     // B: K 0..15 vs 16..31 contiguous
  const int NCH = NPAD / 64;

  stage_panel(gpanel, panel[0], tid);                  // prologue: panel 0
  for (int ccn = 0; ccn < NCH; ++ccn) {
    if (ccn + 1 < NCH)                                 // stream next panel
      stage_panel(gpanel + (size_t)(ccn + 1) * 64 * CC, panel[(ccn + 1) & 1], tid);
#if HAVE_ASYNC
    if (ccn + 1 < NCH) __builtin_amdgcn_s_wait_asynccnt(4);  // panel ccn done
    else               __builtin_amdgcn_s_wait_asynccnt(0);
#endif
    if (tid < 64) sqcol[tid] = sq[abase + ccn * 64 + tid];
    __syncthreads();                                   // panel visible to all waves
    const unsigned short* lb = panel[ccn & 1];
    v8f acc0 = {}, acc1 = {}, acc2 = {}, acc3 = {};
    #pragma unroll
    for (int ks = 0; ks < 4; ++ks) {
      const unsigned short* ap = arow + ks * 32;
      union { v16us v; v8us h[2]; } ua;
      ua.h[0] = *(const v8us*)(ap + o0);
      ua.h[1] = *(const v8us*)(ap + o0 + 8);
      v16bf A = as_bf16(ua.v);
      #pragma unroll
      for (int c = 0; c < 4; ++c) {
        const unsigned short* bp = lb + (c * 16 + (lane & 15)) * PSTRIDE + ks * 32 + bo;
        v16bf Bv = as_bf16(*(const v16us*)bp);
        v8f* pa = (c == 0) ? &acc0 : (c == 1) ? &acc1 : (c == 2) ? &acc2 : &acc3;
        *pa = __builtin_amdgcn_wmma_f32_16x16x32_bf16(false, A, false, Bv,
                                                      (short)0, *pa, false, false);
      }
    }
    #pragma unroll
    for (int c = 0; c < 4; ++c) {
      v8f acc = (c == 0) ? acc0 : (c == 1) ? acc1 : (c == 2) ? acc2 : acc3;
      int colL = c * 16 + (lane & 15);
      float sc = sqcol[colL];
      #pragma unroll
      for (int r = 0; r < 8; ++r) {
        int rowL = wave * 16 + ((lane < 16) ? r : r + 8);
        pdT[rowL * 65 + colL] = 2.f * acc[r] - sqrow[rowL] - sc;
      }
    }
    __syncthreads();
    if (tid < 128) {
      for (int j = 0; j < 64; ++j) {
        float v = pdT[tid * 65 + j];
        if (v > tv[KNN - 1]) {                 // strict > : ties keep lower idx
          int p = KNN - 1;
          while (p > 0 && tv[p - 1] < v) { tv[p] = tv[p - 1]; ti[p] = ti[p - 1]; --p; }
          tv[p] = v; ti[p] = ccn * 64 + j;
        }
      }
    }
    __syncthreads();   // all reads of panel[(ccn+1)&1] (from iter ccn-1) are done
  }
  if (tid < 128) {
    int row = row0 + tid;
    if (row < NN) {
      int* dst = idxout + ((size_t)(a * BB + b) * NN + row) * KNN;
      for (int k = 0; k < KNN; ++k) dst[k] = ti[k];
    }
  }
}

// ---------------------------------------------------------------------------
// conv1 as GEMM: y1[br,b,co,n,t] = sum_{dk,ci} W[co,ci,dk]*feat[ci, 3t+dk].
// Block = 128co x 64n for one (br,b,ntile); graph feature gathered on the fly
// into an LDS [n][k] bf16 panel per 32-deep K chunk (24 chunks x 3 t).
// ---------------------------------------------------------------------------
__global__ void __launch_bounds__(256) conv1_kernel(const float* __restrict__ x,
                const int* __restrict__ idxbuf, const unsigned short* __restrict__ w1blob,
                const float* __restrict__ b1_0, const float* __restrict__ b1_1,
                float* __restrict__ y1) {
  __shared__ unsigned short ldsB[64 * 48];
  int tid = threadIdx.x, wave = tid >> 5, lane = tid & 31;
  int bx = blockIdx.x;
  int nt = bx & 31, b = (bx >> 5) & 7, br = bx >> 8;
  const int* idxa = idxbuf + (size_t)br * BB * NN * KNN;
  const unsigned short* wb = w1blob + (size_t)br * 3 * 8 * 8 * 512;
  const float* bias = br ? b1_1 : b1_0;
  int bo = (lane < 16) ? 0 : 16;
  for (int t = 0; t < 3; ++t) {
    v8f acc0 = {}, acc1 = {}, acc2 = {}, acc3 = {};
    for (int dk = 0; dk < 3; ++dk) {
      for (int cib = 0; cib < 8; ++cib) {
        __syncthreads();
        #pragma unroll
        for (int e = 0; e < 8; ++e) {
          int flat = tid * 8 + e;
          int kk = flat & 31, nn2 = flat >> 5;
          int n = nt * 64 + nn2; if (n > NN - 1) n = NN - 1;
          int ci = cib * 32 + kk;
          float v;
          if (ci < CC) {
            v = x[(b * CC + ci) * NN + n];                       // center
          } else {
            int cim = ci - CC;
            int nb = idxa[(b * NN + n) * KNN + 3 * t + dk];
            v = x[(b * CC + cim) * NN + n] - x[(b * CC + cim) * NN + nb];
          }
          ldsB[nn2 * 48 + kk] = f2bf(v);
        }
        __syncthreads();
        const unsigned short* ap = wb + ((size_t)((dk * 8 + cib) * 8 + wave)) * 512 + lane * 16;
        v16bf A = as_bf16(*(const v16us*)ap);
        #pragma unroll
        for (int c = 0; c < 4; ++c) {
          const unsigned short* bp = &ldsB[(c * 16 + (lane & 15)) * 48 + bo];
          v16bf Bv = as_bf16(*(const v16us*)bp);
          v8f* pa = (c == 0) ? &acc0 : (c == 1) ? &acc1 : (c == 2) ? &acc2 : &acc3;
          *pa = __builtin_amdgcn_wmma_f32_16x16x32_bf16(false, A, false, Bv,
                                                        (short)0, *pa, false, false);
        }
      }
    }
    #pragma unroll
    for (int c = 0; c < 4; ++c) {
      v8f acc = (c == 0) ? acc0 : (c == 1) ? acc1 : (c == 2) ? acc2 : acc3;
      #pragma unroll
      for (int r = 0; r < 8; ++r) {
        int co = wave * 16 + ((lane < 16) ? r : r + 8);
        int n = nt * 64 + c * 16 + (lane & 15);
        if (n < NN)
          y1[(((size_t)(br * BB + b) * CC + co) * NN + n) * 3 + t] = acc[r] + bias[co];
      }
    }
  }
}

// Per-channel batch-norm stats -> fused scale/shift. L=6000 for y1, 2000 for y2.
__global__ void __launch_bounds__(256) bnstats_kernel(const float* __restrict__ y,
                const float* __restrict__ g0, const float* __restrict__ be0,
                const float* __restrict__ g1, const float* __restrict__ be1,
                float* __restrict__ scale, float* __restrict__ shift, int L) {
  __shared__ float s1s[256], s2s[256];
  int br = blockIdx.x >> 7, ci = blockIdx.x & 127;
  size_t base = ((size_t)br * BB * CC + ci) * L;
  float s1 = 0.f, s2 = 0.f;
  for (int b = 0; b < BB; ++b) {
    const float* p = y + base + (size_t)b * CC * L;
    for (int i = threadIdx.x; i < L; i += 256) { float v = p[i]; s1 += v; s2 += v * v; }
  }
  s1s[threadIdx.x] = s1; s2s[threadIdx.x] = s2;
  __syncthreads();
  for (int s = 128; s > 0; s >>= 1) {
    if (threadIdx.x < s) { s1s[threadIdx.x] += s1s[threadIdx.x + s];
                           s2s[threadIdx.x] += s2s[threadIdx.x + s]; }
    __syncthreads();
  }
  if (threadIdx.x == 0) {
    float M = (float)BB * (float)L;
    float mean = s1s[0] / M;
    float var = s2s[0] / M - mean * mean;
    const float* g = br ? g1 : g0;  const float* be = br ? be1 : be0;
    float sc = g[ci] * rsqrtf(var + 1e-5f);
    scale[br * CC + ci] = sc;
    shift[br * CC + ci] = be[ci] - mean * sc;
  }
}

// conv2 GEMM (K=384): BN+ReLU of y1 fused into the bf16 operand staging.
__global__ void __launch_bounds__(256) conv2_kernel(const float* __restrict__ y1,
                const unsigned short* __restrict__ w2blob,
                const float* __restrict__ scale1, const float* __restrict__ shift1,
                const float* __restrict__ b2_0, const float* __restrict__ b2_1,
                float* __restrict__ y2) {
  __shared__ unsigned short ldsB[64 * 48];
  int tid = threadIdx.x, wave = tid >> 5, lane = tid & 31;
  int bx = blockIdx.x;
  int nt = bx & 31, b = (bx >> 5) & 7, br = bx >> 8;
  const unsigned short* wb = w2blob + (size_t)br * 12 * 8 * 512;
  const float* bias = br ? b2_1 : b2_0;
  int bo = (lane < 16) ? 0 : 16;
  v8f acc0 = {}, acc1 = {}, acc2 = {}, acc3 = {};
  for (int s = 0; s < 12; ++s) {
    __syncthreads();
    #pragma unroll
    for (int e = 0; e < 8; ++e) {
      int flat = tid * 8 + e;
      int kk = flat & 31, nn2 = flat >> 5;
      int n = nt * 64 + nn2; if (n > NN - 1) n = NN - 1;
      int k = s * 32 + kk;
      int dk = k >> 7, ci = k & 127;
      float v = y1[(((size_t)(br * BB + b) * CC + ci) * NN + n) * 3 + dk];
      v = v * scale1[br * CC + ci] + shift1[br * CC + ci];
      v = v > 0.f ? v : 0.f;
      ldsB[nn2 * 48 + kk] = f2bf(v);
    }
    __syncthreads();
    const unsigned short* ap = wb + ((size_t)(s * 8 + wave)) * 512 + lane * 16;
    v16bf A = as_bf16(*(const v16us*)ap);
    #pragma unroll
    for (int c = 0; c < 4; ++c) {
      const unsigned short* bp = &ldsB[(c * 16 + (lane & 15)) * 48 + bo];
      v16bf Bv = as_bf16(*(const v16us*)bp);
      v8f* pa = (c == 0) ? &acc0 : (c == 1) ? &acc1 : (c == 2) ? &acc2 : &acc3;
      *pa = __builtin_amdgcn_wmma_f32_16x16x32_bf16(false, A, false, Bv,
                                                    (short)0, *pa, false, false);
    }
  }
  #pragma unroll
  for (int c = 0; c < 4; ++c) {
    v8f acc = (c == 0) ? acc0 : (c == 1) ? acc1 : (c == 2) ? acc2 : acc3;
    #pragma unroll
    for (int r = 0; r < 8; ++r) {
      int co = wave * 16 + ((lane < 16) ? r : r + 8);
      int n = nt * 64 + c * 16 + (lane & 15);
      if (n < NN)
        y2[((size_t)(br * BB + b) * CC + co) * NN + n] = acc[r] + bias[co];
    }
  }
}

// out = relu(bn(y2_br0)) + delta * relu(bn(y2_br1))
__global__ void final_kernel(const float* __restrict__ y2,
                             const float* __restrict__ scale2, const float* __restrict__ shift2,
                             const float* __restrict__ delta, float* __restrict__ out) {
  size_t i = (size_t)blockIdx.x * blockDim.x + threadIdx.x;
  size_t total = (size_t)BB * CC * NN;
  if (i >= total) return;
  int co = (int)((i / NN) % CC);
  float v0 = y2[i] * scale2[co] + shift2[co];                     v0 = v0 > 0.f ? v0 : 0.f;
  float v1 = y2[total + i] * scale2[CC + co] + shift2[CC + co];   v1 = v1 > 0.f ? v1 : 0.f;
  out[i] = v0 + delta[0] * v1;
}

extern "C" void kernel_launch(void* const* d_in, const int* in_sizes, int n_in,
                              void* d_out, int out_size, void* d_ws, size_t ws_size,
                              hipStream_t stream) {
  const float* features = (const float*)d_in[0];
  const float* motion   = (const float*)d_in[1];
  const float* d1_w1 = (const float*)d_in[2];  const float* d1_b1  = (const float*)d_in[3];
  const float* d1_g1 = (const float*)d_in[4];  const float* d1_be1 = (const float*)d_in[5];
  const float* d1_w2 = (const float*)d_in[6];  const float* d1_b2  = (const float*)d_in[7];
  const float* d1_g2 = (const float*)d_in[8];  const float* d1_be2 = (const float*)d_in[9];
  const float* d2_w1 = (const float*)d_in[10]; const float* d2_b1  = (const float*)d_in[11];
  const float* d2_g1 = (const float*)d_in[12]; const float* d2_be1 = (const float*)d_in[13];
  const float* d2_w2 = (const float*)d_in[14]; const float* d2_b2  = (const float*)d_in[15];
  const float* d2_g2 = (const float*)d_in[16]; const float* d2_be2 = (const float*)d_in[17];
  const float* delta = (const float*)d_in[18];

  char* ws = (char*)d_ws;
  size_t off = 0;
  auto alloc = [&](size_t bytes) { size_t o = off; off += (bytes + 255) & ~(size_t)255; return o; };
  unsigned short* xb     = (unsigned short*)(ws + alloc((size_t)2 * BB * NPAD * CC * 2));
  float*          sq     = (float*)(ws + alloc((size_t)2 * BB * NPAD * 4));
  int*            idxbuf = (int*)(ws + alloc((size_t)2 * BB * NN * KNN * 4));
  unsigned short* w1blob = (unsigned short*)(ws + alloc((size_t)2 * 3 * 8 * 8 * 512 * 2));
  unsigned short* w2blob = (unsigned short*)(ws + alloc((size_t)2 * 12 * 8 * 512 * 2));
  float*          y1     = (float*)(ws + alloc((size_t)2 * BB * CC * NN * 3 * 4));
  float*          y2     = (float*)(ws + alloc((size_t)2 * BB * CC * NN * 4));
  float*          scale1 = (float*)(ws + alloc(2 * CC * 4));
  float*          shift1 = (float*)(ws + alloc(2 * CC * 4));
  float*          scale2 = (float*)(ws + alloc(2 * CC * 4));
  float*          shift2 = (float*)(ws + alloc(2 * CC * 4));

  prep_kernel  <<<(2 * BB * NPAD) / 256, 256, 0, stream>>>(features, motion, xb, sq);
  wprep1_kernel<<<(2 * 3 * 8 * 8 * 512) / 256, 256, 0, stream>>>(d1_w1, d2_w1, w1blob);
  wprep2_kernel<<<(2 * 12 * 8 * 512) / 256, 256, 0, stream>>>(d1_w2, d2_w2, w2blob);
  gram_topk_kernel<<<2 * BB * (NPAD / 128), 256, 0, stream>>>(xb, sq, idxbuf);
  conv1_kernel <<<2 * BB * 32, 256, 0, stream>>>(features, idxbuf, w1blob, d1_b1, d2_b1, y1);
  bnstats_kernel<<<2 * CC, 256, 0, stream>>>(y1, d1_g1, d1_be1, d2_g1, d2_be1, scale1, shift1, NN * 3);
  conv2_kernel <<<2 * BB * 32, 256, 0, stream>>>(y1, w2blob, scale1, shift1, d1_b2, d2_b2, y2);
  bnstats_kernel<<<2 * CC, 256, 0, stream>>>(y2, d1_g2, d1_be2, d2_g2, d2_be2, scale2, shift2, NN);
  final_kernel <<<(unsigned)(((size_t)BB * CC * NN + 255) / 256), 256, 0, stream>>>(
      y2, scale2, shift2, delta, (float*)d_out);
  (void)in_sizes; (void)n_in; (void)out_size; (void)ws_size;
}
